// BlockAdaptiveTransformerLayer_77008763617332
// MI455X (gfx1250) — matrix operand order repaired and checked by reading, not compiled, not run on hardware
//
#include <hip/hip_runtime.h>
#include <math.h>

#define N_NODES 4096
#define L_SEQ 8
#define DS 128
#define DV 32
#define NH 8
#define DHID 128
#define E_EDGES 32768
#define DSH 16
#define DVH 4
#define QDIM 28 /* DSH + DVH*3 */

// f16 weight pool offsets (in halves) inside workspace, after q/k buffers
#define W1QK_OFF 0       /* 160x128 */
#define W2QK_OFF 20480   /* 128x48  */
#define W1V_OFF  26624   /* 160x128 */
#define W2V_OFF  47104   /* 128x32  */
#define WPOOL_HALVES 51200

typedef __attribute__((ext_vector_type(16))) _Float16 v16h;
typedef __attribute__((ext_vector_type(8)))  float    v8f;

__device__ __forceinline__ float siluf(float x) { return x / (1.f + __expf(-x)); }
__device__ __forceinline__ float sclamp(float x) { return (fabsf(x) < 1e-10f) ? 1e-10f : x; }

// ---- CDNA5 async global->LDS staging (ASYNCcnt path, ISA 15.18.3 op 98) ----
// Single static __shared__ block => LDS byte offsets are absolute offsets.
__device__ __forceinline__ void stage16(char* smem, unsigned lds_off,
                                        const char* gbase, unsigned goff) {
  (void)smem;
  asm volatile("global_load_async_to_lds_b128 %0, %1, %2"
               :
               : "v"(lds_off), "v"(goff), "s"(gbase)
               : "memory");
}
__device__ __forceinline__ void stage_wait() {
  asm volatile("s_wait_asynccnt 0" ::: "memory");
}

// A fragment: 16x32 f16 tile, row-major in LDS with row stride `ld` (halves).
// lanes 0-15 M=0..15 hold K=0..15 (2 per VGPR), lanes 16-31 hold K=16..31.
__device__ __forceinline__ v16h load_A_frag(const _Float16* p, int lane, int ld) {
  const int g = (lane >> 4) & 1, m = lane & 15;
  v16h a;
#pragma unroll
  for (int v = 0; v < 8; ++v) {
    int k = 2 * v + g * 16;
    a[2 * v]     = p[m * ld + k];
    a[2 * v + 1] = p[m * ld + k + 1];
  }
  return a;
}

// B fragment: 32x16 f16 tile, row-major [K][N] in LDS with row stride `ld`.
// lanes 0-15 hold K=0..15 (2 per VGPR), lanes 16-31 hold K=16..31.
__device__ __forceinline__ v16h load_B_frag(const _Float16* p, int lane, int ld) {
  const int g = (lane >> 4) & 1, n = lane & 15;
  v16h b;
#pragma unroll
  for (int v = 0; v < 8; ++v) {
    int k = 2 * v + g * 16;
    b[2 * v]     = p[k * ld + n];
    b[2 * v + 1] = p[(k + 1) * ld + n];
  }
  return b;
}

__device__ __forceinline__ v8f wmma16(v16h a, v16h b, v8f c) {
  return __builtin_amdgcn_wmma_f32_16x16x32_f16(false, a, false, b, (short)0, c,
                                                false, false);
}

// ---------------------------------------------------------------------------
// Kernel W: convert f32 weights -> padded f16 in workspace
// ---------------------------------------------------------------------------
__global__ void convert_pad_f16(const float* __restrict__ src,
                                _Float16* __restrict__ dst, int rs, int cs,
                                int rp, int cp) {
  int i = blockIdx.x * 256 + threadIdx.x;
  if (i >= rp * cp) return;
  int r = i / cp, c = i % cp;
  dst[i] = (r < rs && c < cs) ? (_Float16)src[r * cs + c] : (_Float16)0.f;
}

// ---------------------------------------------------------------------------
// Kernel 0: initialize d_out = [H | V | X] (edge kernel accumulates on top)
// ---------------------------------------------------------------------------
__global__ void init_out_kernel(const float* __restrict__ H,
                                const float* __restrict__ V,
                                const float* __restrict__ X,
                                float* __restrict__ out) {
  long long i = (long long)blockIdx.x * blockDim.x + threadIdx.x;
  const long long nh = (long long)N_NODES * L_SEQ * DS;
  const long long nv = (long long)N_NODES * L_SEQ * DV * 3;
  const long long nx = (long long)N_NODES * L_SEQ * 3;
  if (i < nh) out[i] = H[i];
  else if (i < nh + nv) out[i] = V[i - nh];
  else if (i < nh + nv + nx) out[i] = X[i - nh - nv];
}

// ---------------------------------------------------------------------------
// Kernel 1: node GVP (q/k projection). One block per node = 64 tokens (h,l).
// ---------------------------------------------------------------------------
__global__ __launch_bounds__(256) void node_gvp_qk_kernel(
    const float* __restrict__ H, const float* __restrict__ V,
    const float* __restrict__ Wv, const _Float16* __restrict__ W1h,
    const float* __restrict__ b1, const _Float16* __restrict__ W2h,
    const float* __restrict__ b2, float* __restrict__ qbuf,
    float* __restrict__ kbuf) {
  __shared__ __align__(16) char smem[51200];
  _Float16* sA   = (_Float16*)smem;            // 64x160 f16 s-matrix    [0,20480)
  float*    sS2  = (float*)smem;               // 64x48 f32 (alias late)
  _Float16* sW   = (_Float16*)(smem + 20480);  // weight K-chunk         [20480,28672)
  _Float16* sAct = (_Float16*)(smem + 28672);  // 64x128 f16 activations [28672,45056)
  float*    sV2  = (float*)(smem + 45056);     // 64 x [c*8+k]

  const int node = blockIdx.x;
  const int tid  = threadIdx.x;
  const int wave = tid >> 5, lane = tid & 31;

  // ---- build s rows: token t = h*8+l, 4 threads per token ----
  const int t = tid >> 2, sub = tid & 3;
  const int h = t >> 3, l = t & 7;
  if (sub == 0) {
    const float4* hp = (const float4*)(H + (node * L_SEQ + l) * DS + h * DSH);
#pragma unroll
    for (int qv = 0; qv < 4; ++qv) {
      float4 hv = hp[qv];
      sA[t * 160 + qv * 4 + 0] = (_Float16)hv.x;
      sA[t * 160 + qv * 4 + 1] = (_Float16)hv.y;
      sA[t * 160 + qv * 4 + 2] = (_Float16)hv.z;
      sA[t * 160 + qv * 4 + 3] = (_Float16)hv.w;
    }
#pragma unroll
    for (int d = 144; d < 160; ++d) sA[t * 160 + d] = (_Float16)0.f;
  }
  {
    // Vcat(3x8) = [Vt(3x4) | cross(col_i, col_{(i+1)%4})]
    float vc[3][8];
#pragma unroll
    for (int k = 0; k < 4; ++k)
#pragma unroll
      for (int c = 0; c < 3; ++c)
        vc[c][k] = V[((node * L_SEQ + l) * DV + (h * DVH + k)) * 3 + c];
#pragma unroll
    for (int k = 0; k < 4; ++k) {
      int k2 = (k + 1) & 3;
      vc[0][4 + k] = vc[1][k] * vc[2][k2] - vc[2][k] * vc[1][k2];
      vc[1][4 + k] = vc[2][k] * vc[0][k2] - vc[0][k] * vc[2][k2];
      vc[2][4 + k] = vc[0][k] * vc[1][k2] - vc[1][k] * vc[0][k2];
    }
    for (int j = sub; j < 136; j += 4) {  // Vp = Vcat @ Wv_qk(8x136)
      float p0 = 0.f, p1 = 0.f, p2 = 0.f;
#pragma unroll
      for (int i = 0; i < 8; ++i) {
        float w = Wv[i * 136 + j];
        p0 += vc[0][i] * w; p1 += vc[1][i] * w; p2 += vc[2][i] * w;
      }
      if (j < 128) {
        float a0 = sclamp(p0), a1 = sclamp(p1), a2 = sclamp(p2);
        sA[t * 160 + 16 + j] = (_Float16)sqrtf(a0 * a0 + a1 * a1 + a2 * a2);
      } else {
        int k = j - 128;
        sV2[t * 24 + 0 * 8 + k] = p0;
        sV2[t * 24 + 1 * 8 + k] = p1;
        sV2[t * 24 + 2 * 8 + k] = p2;
      }
    }
  }
  __syncthreads();

  // ---- GEMM1: 64x160 @ W1_qk(padded 160x128 f16) ----
  v8f acc[4] = {{0.f,0.f,0.f,0.f,0.f,0.f,0.f,0.f},
                {0.f,0.f,0.f,0.f,0.f,0.f,0.f,0.f},
                {0.f,0.f,0.f,0.f,0.f,0.f,0.f,0.f},
                {0.f,0.f,0.f,0.f,0.f,0.f,0.f,0.f}};
  for (int kt = 0; kt < 5; ++kt) {
    // async-stage 32x128 f16 chunk (8192 B) into sW
    for (int u = tid; u < 512; u += 256)
      stage16(smem, 20480u + u * 16, (const char*)W1h + kt * 8192, u * 16);
    stage_wait();
    __syncthreads();
    v16h bf = load_B_frag(sW + wave * 16, lane, 128);
#pragma unroll
    for (int mt = 0; mt < 4; ++mt) {
      v16h af = load_A_frag(sA + (mt * 16) * 160 + kt * 32, lane, 160);
      acc[mt] = wmma16(af, bf, acc[mt]);
    }
    __syncthreads();
  }
  {  // bias + silu -> sAct (f16)
    const int g = (lane >> 4) & 1, nid = lane & 15;
    const int ncol = wave * 16 + nid;
    const float bb = b1[ncol];
#pragma unroll
    for (int mt = 0; mt < 4; ++mt)
#pragma unroll
      for (int r = 0; r < 8; ++r) {
        int M = mt * 16 + g * 8 + r;
        sAct[M * 128 + ncol] = (_Float16)siluf(acc[mt][r] + bb);
      }
  }
  __syncthreads();

  // ---- GEMM2: 64x128 @ W2_qk(padded 128x48 f16); 12 tiles / 8 waves ----
  v8f acc2[2] = {{0.f,0.f,0.f,0.f,0.f,0.f,0.f,0.f},
                 {0.f,0.f,0.f,0.f,0.f,0.f,0.f,0.f}};
  for (int kt = 0; kt < 4; ++kt) {
    for (int u = tid; u < 192; u += 256)  // 32x48 f16 = 3072 B
      stage16(smem, 20480u + u * 16, (const char*)W2h + kt * 3072, u * 16);
    stage_wait();
    __syncthreads();
#pragma unroll
    for (int ti = 0; ti < 2; ++ti) {
      int tt = wave + ti * 8;
      if (tt < 12) {  // wave-uniform branch: EXEC stays all-ones for WMMA
        int mt = tt & 3, nt = tt >> 2;
        v16h bf = load_B_frag(sW + nt * 16, lane, 48);
        v16h af = load_A_frag(sAct + (mt * 16) * 128 + kt * 32, lane, 128);
        acc2[ti] = wmma16(af, bf, acc2[ti]);
      }
    }
    __syncthreads();
  }
  {  // write s2 (aliases sA region; sA no longer read)
    const int g = (lane >> 4) & 1, nid = lane & 15;
#pragma unroll
    for (int ti = 0; ti < 2; ++ti) {
      int tt = wave + ti * 8;
      if (tt < 12) {
        int mt = tt & 3, nt = tt >> 2;
        int ncol = nt * 16 + nid;
        float bb = (ncol < 40) ? b2[ncol] : 0.f;
#pragma unroll
        for (int r = 0; r < 8; ++r) {
          int M = mt * 16 + g * 8 + r;
          sS2[M * 48 + ncol] = acc2[ti][r] + bb;
        }
      }
    }
  }
  __syncthreads();

  // ---- emit q / kk (28 dims each) ----
  for (int idx = tid; idx < 64 * 56; idx += 256) {
    int tt = idx / 56, d2 = idx % 56;
    long long base = ((long long)node * NH * L_SEQ + tt) * QDIM;
    if (d2 < QDIM) {
      float v;
      if (d2 < 16) v = sS2[tt * 48 + d2];
      else {
        int k = (d2 - 16) / 3, c = (d2 - 16) % 3;
        v = sS2[tt * 48 + 32 + k] * sV2[tt * 24 + c * 8 + k];
      }
      qbuf[base + d2] = v;
    } else {
      int d = d2 - QDIM;
      float v;
      if (d < 16) v = sS2[tt * 48 + 16 + d];
      else {
        int k = (d - 16) / 3 + 4, c = (d - 16) % 3;
        v = sS2[tt * 48 + 32 + k] * sV2[tt * 24 + c * 8 + k];
      }
      kbuf[base + d] = v;
    }
  }
}

// ---------------------------------------------------------------------------
// Kernel 2: fused per-edge attention + GVP. One block per edge, 64 tokens.
// Keeps R (268 MB) and alpha (67 MB) in LDS instead of HBM.
// ---------------------------------------------------------------------------
__global__ __launch_bounds__(256) void edge_attn_gvp_kernel(
    const float* __restrict__ H, const float* __restrict__ V,
    const float* __restrict__ X, const int* __restrict__ ei,
    const float* __restrict__ Wv, const _Float16* __restrict__ W1h,
    const float* __restrict__ b1, const _Float16* __restrict__ W2h,
    const float* __restrict__ b2, const float* __restrict__ WR,
    const float* __restrict__ qbuf, const float* __restrict__ kbuf,
    float* __restrict__ outH, float* __restrict__ outV) {
  __shared__ __align__(16) char smem[59904];
  _Float16* sA    = (_Float16*)smem;            // 64x160 f16 s-matrix
  float*    sS2   = (float*)smem;               // 64x32 f32 (alias late)
  _Float16* sW    = (_Float16*)(smem + 20480);  // weight K-chunk
  _Float16* sAct  = (_Float16*)(smem + 28672);  // 64x128 f16 acts (GEMM phase)
  float*    sQ    = (float*)(smem + 28672);     // 1792 f (early phase alias)
  float*    sK    = (float*)(smem + 28672) + 1792;
  float*    sAlpha= (float*)(smem + 45056);     // [head][64 pairs]
  float*    sXij  = (float*)(smem + 47104);     // [64 pairs][3]
  float*    sR    = (float*)(smem + 47872);     // [64 pairs][32 rbf]
  float*    sV2   = (float*)(smem + 56064);     // [64 tok][c*4+k]
  float*    sAggX = (float*)(smem + 59136);     // [64 tok][3]

  const int e   = blockIdx.x;
  const int tid = threadIdx.x;
  const int wave = tid >> 5, lane = tid & 31;
  const int row = ei[e], col = ei[E_EDGES + e];

  // ---- X_ij + stage q[row], kk[col] (vectorized) ----
  for (int idx = tid; idx < 192; idx += 256) {
    int pr = idx / 3, c = idx % 3, li = pr >> 3, lj = pr & 7;
    sXij[idx] = X[(row * L_SEQ + li) * 3 + c] - X[(col * L_SEQ + lj) * 3 + c];
  }
  {
    const float4* qg = (const float4*)(qbuf + (long long)row * (NH * L_SEQ * QDIM));
    const float4* kg = (const float4*)(kbuf + (long long)col * (NH * L_SEQ * QDIM));
    float4* q4 = (float4*)sQ;
    float4* k4 = (float4*)sK;
    for (int idx = tid; idx < 896; idx += 256) {
      if (idx < 448) q4[idx] = qg[idx];
      else k4[idx - 448] = kg[idx - 448];
    }
  }
  __syncthreads();

  // ---- RBF: R[pair][m] (m = head*4 + r after reshape) ----
  const float dstep = 10.0f / 31.0f;
  const float coeff = -0.5f / (dstep * dstep);
  for (int idx = tid; idx < 2048; idx += 256) {
    int pr = idx >> 5, m = idx & 31;
    float x0 = sclamp(sXij[pr * 3 + 0]);
    float x1 = sclamp(sXij[pr * 3 + 1]);
    float x2 = sclamp(sXij[pr * 3 + 2]);
    float D  = sqrtf(x0 * x0 + x1 * x1 + x2 * x2);
    float dd = D - dstep * (float)m;
    sR[idx] = __expf(coeff * dd * dd);
  }
  __syncthreads();

  // ---- scores, gate, alpha = silu(score * gate) ----
  const float wr0 = WR[0], wr1 = WR[1], wr2 = WR[2], wr3 = WR[3];
  for (int idx = tid; idx < 512; idx += 256) {
    int n = idx >> 6, pr = idx & 63, li = pr >> 3, lj = pr & 7;
    const float* qp = sQ + (n * 8 + li) * QDIM;
    const float* kp = sK + (n * 8 + lj) * QDIM;
    float sc = 0.f;
#pragma unroll
    for (int d = 0; d < QDIM; ++d) sc += qp[d] * kp[d];
    const float* rp = sR + pr * 32 + n * 4;
    float g = rp[0] * wr0 + rp[1] * wr1 + rp[2] * wr2 + rp[3] * wr3;
    sAlpha[idx] = siluf(sc * g);
  }
  __syncthreads();

  // ---- token prep part 1 (token t = head*8 + lj) ----
  const int t = tid >> 2, sub = tid & 3;
  const int n = t >> 3, lj = t & 7;
  if (sub == 0) {  // aggX[t][c] = sum_li alpha * X_ij
#pragma unroll
    for (int c = 0; c < 3; ++c) {
      float s = 0.f;
#pragma unroll
      for (int li = 0; li < 8; ++li)
        s += sAlpha[n * 64 + li * 8 + lj] * sXij[(li * 8 + lj) * 3 + c];
      sAggX[t * 3 + c] = s;
    }
  } else if (sub == 1) {  // Hv_in[16:20] = sum_li alpha * R
#pragma unroll
    for (int r = 0; r < 4; ++r) {
      float s = 0.f;
#pragma unroll
      for (int li = 0; li < 8; ++li)
        s += sAlpha[n * 64 + li * 8 + lj] * sR[(li * 8 + lj) * 32 + n * 4 + r];
      sA[t * 160 + 16 + r] = (_Float16)s;
    }
  } else if (sub == 2) {  // Hv_in[0:16] = Hh[col]
    const float4* hp = (const float4*)(H + (col * L_SEQ + lj) * DS + n * DSH);
#pragma unroll
    for (int qv = 0; qv < 4; ++qv) {
      float4 hv = hp[qv];
      sA[t * 160 + qv * 4 + 0] = (_Float16)hv.x;
      sA[t * 160 + qv * 4 + 1] = (_Float16)hv.y;
      sA[t * 160 + qv * 4 + 2] = (_Float16)hv.z;
      sA[t * 160 + qv * 4 + 3] = (_Float16)hv.w;
    }
  } else {  // K padding
#pragma unroll
    for (int d = 148; d < 160; ++d) sA[t * 160 + d] = (_Float16)0.f;
  }
  __syncthreads();

  // ---- token prep part 2: Vcat(3x10) = [Vh(4) | aggX | crosses(5)] ----
  {
    float vc[3][10];
#pragma unroll
    for (int k = 0; k < 4; ++k)
#pragma unroll
      for (int c = 0; c < 3; ++c)
        vc[c][k] = V[((col * L_SEQ + lj) * DV + (n * DVH + k)) * 3 + c];
#pragma unroll
    for (int c = 0; c < 3; ++c) vc[c][4] = sAggX[t * 3 + c];
#pragma unroll
    for (int k = 0; k < 5; ++k) {
      int k2 = (k + 1) % 5;
      vc[0][5 + k] = vc[1][k] * vc[2][k2] - vc[2][k] * vc[1][k2];
      vc[1][5 + k] = vc[2][k] * vc[0][k2] - vc[0][k] * vc[2][k2];
      vc[2][5 + k] = vc[0][k] * vc[1][k2] - vc[1][k] * vc[0][k2];
    }
    for (int j = sub; j < 132; j += 4) {  // Vp = Vcat @ Wv_v(10x132)
      float p0 = 0.f, p1 = 0.f, p2 = 0.f;
#pragma unroll
      for (int i = 0; i < 10; ++i) {
        float w = Wv[i * 132 + j];
        p0 += vc[0][i] * w; p1 += vc[1][i] * w; p2 += vc[2][i] * w;
      }
      if (j < 128) {
        float a0 = sclamp(p0), a1 = sclamp(p1), a2 = sclamp(p2);
        sA[t * 160 + 20 + j] = (_Float16)sqrtf(a0 * a0 + a1 * a1 + a2 * a2);
      } else {
        int k = j - 128;
        sV2[t * 12 + 0 * 4 + k] = p0;
        sV2[t * 12 + 1 * 4 + k] = p1;
        sV2[t * 12 + 2 * 4 + k] = p2;
      }
    }
  }
  __syncthreads();

  // ---- GEMM1: 64x160 @ W1_v(padded 160x128 f16) ----
  v8f acc[4] = {{0.f,0.f,0.f,0.f,0.f,0.f,0.f,0.f},
                {0.f,0.f,0.f,0.f,0.f,0.f,0.f,0.f},
                {0.f,0.f,0.f,0.f,0.f,0.f,0.f,0.f},
                {0.f,0.f,0.f,0.f,0.f,0.f,0.f,0.f}};
  for (int kt = 0; kt < 5; ++kt) {
    for (int u = tid; u < 512; u += 256)
      stage16(smem, 20480u + u * 16, (const char*)W1h + kt * 8192, u * 16);
    stage_wait();
    __syncthreads();
    v16h bf = load_B_frag(sW + wave * 16, lane, 128);
#pragma unroll
    for (int mt = 0; mt < 4; ++mt) {
      v16h af = load_A_frag(sA + (mt * 16) * 160 + kt * 32, lane, 160);
      acc[mt] = wmma16(af, bf, acc[mt]);
    }
    __syncthreads();
  }
  {  // bias + silu -> sAct
    const int g = (lane >> 4) & 1, nid = lane & 15;
    const int ncol = wave * 16 + nid;
    const float bb = b1[ncol];
#pragma unroll
    for (int mt = 0; mt < 4; ++mt)
#pragma unroll
      for (int r = 0; r < 8; ++r) {
        int M = mt * 16 + g * 8 + r;
        sAct[M * 128 + ncol] = (_Float16)siluf(acc[mt][r] + bb);
      }
  }
  __syncthreads();

  // ---- GEMM2: 64x128 @ W2_v(padded 128x32 f16); 8 tiles / 8 waves ----
  v8f acc2 = {0.f,0.f,0.f,0.f,0.f,0.f,0.f,0.f};
  const int mt2 = wave & 3, nt2 = wave >> 2;
  for (int kt = 0; kt < 4; ++kt) {
    for (int u = tid; u < 128; u += 256)  // 32x32 f16 = 2048 B
      stage16(smem, 20480u + u * 16, (const char*)W2h + kt * 2048, u * 16);
    stage_wait();
    __syncthreads();
    v16h bf = load_B_frag(sW + nt2 * 16, lane, 32);
    v16h af = load_A_frag(sAct + (mt2 * 16) * 128 + kt * 32, lane, 128);
    acc2 = wmma16(af, bf, acc2);
    __syncthreads();
  }
  {  // write s2 f32 (aliases sA region; no longer read)
    const int g = (lane >> 4) & 1, nid = lane & 15;
    const int ncol = nt2 * 16 + nid;
    const float bb = (ncol < 20) ? b2[ncol] : 0.f;
#pragma unroll
    for (int r = 0; r < 8; ++r) {
      int M = mt2 * 16 + g * 8 + r;
      sS2[M * 32 + ncol] = acc2[r] + bb;
    }
  }
  __syncthreads();

  // ---- alpha-weighted aggregation over lj, scatter-add into node `row` ----
  for (int idx = tid; idx < 1024; idx += 256) {  // H: 64 (n,li) x 16
    int nl = idx >> 4, d = idx & 15;
    int nn = nl >> 3, li = nl & 7;
    float s = 0.f;
#pragma unroll
    for (int jj = 0; jj < 8; ++jj)
      s += sAlpha[nn * 64 + li * 8 + jj] * sS2[(nn * 8 + jj) * 32 + d];
    atomicAdd(&outH[((long long)row * L_SEQ + li) * DS + nn * DSH + d], s);
  }
  for (int idx = tid; idx < 768; idx += 256) {  // V: 64 (n,li) x (4x3)
    int nl = idx / 12, kc = idx % 12, k = kc / 3, c = kc % 3;
    int nn = nl >> 3, li = nl & 7;
    float s = 0.f;
#pragma unroll
    for (int jj = 0; jj < 8; ++jj) {
      int tj = nn * 8 + jj;
      s += sAlpha[nn * 64 + li * 8 + jj] * sS2[tj * 32 + 16 + k] *
           sV2[tj * 12 + c * 4 + k];
    }
    atomicAdd(&outV[(((long long)row * L_SEQ + li) * DV + nn * DVH + k) * 3 + c], s);
  }
}

// ---------------------------------------------------------------------------
extern "C" void kernel_launch(void* const* d_in, const int* in_sizes, int n_in,
                              void* d_out, int out_size, void* d_ws,
                              size_t ws_size, hipStream_t stream) {
  (void)in_sizes; (void)n_in; (void)out_size; (void)ws_size;
  const float* H     = (const float*)d_in[0];
  const float* V     = (const float*)d_in[1];
  const float* X     = (const float*)d_in[2];
  const int*   ei    = (const int*)d_in[4];
  const float* Wv_qk = (const float*)d_in[5];
  const float* W1_qk = (const float*)d_in[6];
  const float* b1_qk = (const float*)d_in[7];
  const float* W2_qk = (const float*)d_in[8];
  const float* b2_qk = (const float*)d_in[9];
  const float* Wv_v  = (const float*)d_in[10];
  const float* W1_v  = (const float*)d_in[11];
  const float* b1_v  = (const float*)d_in[12];
  const float* W2_v  = (const float*)d_in[13];
  const float* b2_v  = (const float*)d_in[14];
  const float* W_R   = (const float*)d_in[15];

  float* out  = (float*)d_out;
  float* outH = out;
  float* outV = out + (long long)N_NODES * L_SEQ * DS;
  float* qbuf = (float*)d_ws;
  float* kbuf = qbuf + (long long)N_NODES * NH * L_SEQ * QDIM;
  _Float16* wpool = (_Float16*)(kbuf + (long long)N_NODES * NH * L_SEQ * QDIM);

  // pre-convert weights to padded f16
  convert_pad_f16<<<(160 * 128 + 255) / 256, 256, 0, stream>>>(
      W1_qk, wpool + W1QK_OFF, 144, 128, 160, 128);
  convert_pad_f16<<<(128 * 48 + 255) / 256, 256, 0, stream>>>(
      W2_qk, wpool + W2QK_OFF, 128, 40, 128, 48);
  convert_pad_f16<<<(160 * 128 + 255) / 256, 256, 0, stream>>>(
      W1_v, wpool + W1V_OFF, 148, 128, 160, 128);
  convert_pad_f16<<<(128 * 32 + 255) / 256, 256, 0, stream>>>(
      W2_v, wpool + W2V_OFF, 128, 20, 128, 32);

  const long long tot = (long long)N_NODES * L_SEQ * (DS + DV * 3 + 3);
  init_out_kernel<<<(int)((tot + 255) / 256), 256, 0, stream>>>(H, V, X, out);
  node_gvp_qk_kernel<<<N_NODES, 256, 0, stream>>>(
      H, V, Wv_qk, wpool + W1QK_OFF, b1_qk, wpool + W2QK_OFF, b2_qk, qbuf, kbuf);
  edge_attn_gvp_kernel<<<E_EDGES, 256, 0, stream>>>(
      H, V, X, ei, Wv_v, wpool + W1V_OFF, b1_v, wpool + W2V_OFF, b2_v, W_R,
      qbuf, kbuf, outH, outV);
}
